// ShallowLSTM_57097295233424
// MI455X (gfx1250) — compile-verified, matrix operand order checked by reading, not run on hardware
//
#include <hip/hip_runtime.h>

typedef __attribute__((ext_vector_type(16))) _Float16 v16h;
typedef __attribute__((ext_vector_type(8)))  _Float16 v8h;
typedef __attribute__((ext_vector_type(8)))  float    v8f;

#define LSTM_B 8192
#define LSTM_T 180
#define LSTM_I 7
#define LSTM_H 64
#define LSTM_G 256   // 4*H
#define LSTM_O 16
#define WAVES_PER_WG 2
#define ROWS_PER_WAVE 16

// Fast activation math on the v_rcp_f32 / v_exp_f32 path (no IEEE div sequences).
// Limits are exact: exp2(+-inf) and rcp(inf) saturate, so no clamp needed.
__device__ __forceinline__ float sigmoid_f(float xx) {
  return __builtin_amdgcn_rcpf(1.0f + __builtin_amdgcn_exp2f(xx * -1.44269504f));
}
__device__ __forceinline__ float tanh_f(float xx) {
  const float e2 = __builtin_amdgcn_exp2f(xx * 2.88539008f); // e^(2x)
  return 1.0f - 2.0f * __builtin_amdgcn_rcpf(e2 + 1.0f);
}

__global__ __launch_bounds__(WAVES_PER_WG * 32)
void ShallowLSTM_gfx1250_kernel(const float* __restrict__ x,
                                const float* __restrict__ W_ih,
                                const float* __restrict__ W_hh,
                                const float* __restrict__ b_ih,
                                const float* __restrict__ b_hh,
                                const float* __restrict__ W_fc,
                                const float* __restrict__ b_fc,
                                float* __restrict__ out) {
  // 48 B-fragments (3 K-chunks x 16 N-tiles), each 32 lanes x 16 f16 = 1KB.
  // K'=0..63 -> W_hh^T, 64..70 -> W_ih^T, 71 -> combined gate bias (A supplies
  // a constant 1.0 at K'=71), 72..95 -> zero pad.
  __shared__ __align__(32) _Float16 s_wfrag[48 * 32 * 16];   // 49152 B
  __shared__ __align__(16) _Float16 s_h[WAVES_PER_WG][ROWS_PER_WAVE * LSTM_H]; // 2x2048 B

  const int tid  = threadIdx.x;
  const int wave = tid >> 5;
  const int l    = tid & 31;      // lane in wave32
  const int n16  = l & 15;
  const int h16  = (l >> 4) & 1;  // half-wave (K sub-block selector)

  // ---- init: prepack [W_hh^T | W_ih^T | bias | 0] (K=96) in WMMA B-frag order ----
  // B element B[K', N]; per-lane v16h j: K'' = (j<8 ? j : j+8) + 8*h16, N = n16
  for (int f = wave; f < 48; f += WAVES_PER_WG) {
    const int kc = f >> 4;          // K chunk 0..2
    const int nt = f & 15;          // N tile 0..15
    const int N  = nt * 16 + n16;   // gate-output column 0..255
    _Float16* dst = &s_wfrag[f * 512 + l * 16];
#pragma unroll
    for (int j = 0; j < 16; ++j) {
      const int kk = kc * 32 + ((j < 8) ? j : j + 8) + 8 * h16;
      float v = 0.0f;
      if (kk < LSTM_H)                      v = W_hh[N * LSTM_H + kk];
      else if (kk < LSTM_H + LSTM_I)        v = W_ih[N * LSTM_I + (kk - LSTM_H)];
      else if (kk == LSTM_H + LSTM_I)       v = b_ih[N] + b_hh[N]; // K'=71: bias row
      dst[j] = (_Float16)v;
    }
  }

  // ---- init: h stage = 0 ----
  _Float16* hs = &s_h[wave][0];
  for (int idx = l; idx < ROWS_PER_WAVE * LSTM_H; idx += 32)
    hs[idx] = (_Float16)0.0f;

  __syncthreads();

  const int b0 = (blockIdx.x * WAVES_PER_WG + wave) * ROWS_PER_WAVE;

  // c state in registers, C/D layout: c[j][r] -> row (r+8*h16), col j*16+n16
  v8f c[4];
#pragma unroll
  for (int j = 0; j < 4; ++j)
#pragma unroll
    for (int r = 0; r < 8; ++r) c[j][r] = 0.0f;

  const float* xrow = x + ((size_t)(b0 + n16) * LSTM_T) * LSTM_I; // lanes 0-15

  for (int t = 0; t < LSTM_T; ++t) {
    asm volatile("s_wait_dscnt 0" ::: "memory"); // previous step's h stores visible

    // ---- A fragments: [h | x,1 | pad], 16-bit A layout ----
    v16h a[3];
#pragma unroll
    for (int kc = 0; kc < 2; ++kc) {
      const int m  = n16;                 // A: lane holds row m = l%16
      const int k0 = kc * 32 + 8 * h16;   // contiguous K bytes in row-major h stage
      const v8h lo = *reinterpret_cast<const v8h*>(&hs[m * LSTM_H + k0]);
      const v8h hi = *reinterpret_cast<const v8h*>(&hs[m * LSTM_H + k0 + 16]);
#pragma unroll
      for (int j = 0; j < 8; ++j) { a[kc][j] = lo[j]; a[kc][j + 8] = hi[j]; }
    }
    {
      v16h a2;
#pragma unroll
      for (int j = 0; j < 16; ++j) a2[j] = (_Float16)0.0f;
      if (l < 16) {  // chunk2: K'' 0..6 = x_t[0..6]; K''=7 = 1.0 (bias lane)
        const float* xp = xrow + (size_t)t * LSTM_I;
#pragma unroll
        for (int j = 0; j < 7; ++j) a2[j] = (_Float16)xp[j];
        a2[7] = (_Float16)1.0f;
      }
      a[2] = a2;
    }

    // ---- per 16-column block j (fully unrolled: c[j] must be a static reg) ----
#pragma unroll
    for (int j = 0; j < 4; ++j) {
      v8f acc[4];
#pragma unroll
      for (int g = 0; g < 4; ++g) {
        const int nt = g * 4 + j;  // N-tile in [0,16): gate g, col block j
        v8f t0 = {};               // folds to inline SRC2=0 in first WMMA
#pragma unroll
        for (int kc = 0; kc < 3; ++kc) {
          const v16h bf = *reinterpret_cast<const v16h*>(
              &s_wfrag[(kc * 16 + nt) * 512 + l * 16]);
          t0 = __builtin_amdgcn_wmma_f32_16x16x32_f16(
              false, a[kc], false, bf, (short)0, t0, false, false);
        }
        acc[g] = t0;
      }
#pragma unroll
      for (int r = 0; r < 8; ++r) {
        const float ig = sigmoid_f(acc[0][r]);
        const float fg = sigmoid_f(acc[1][r]);
        const float gg = tanh_f(acc[2][r]);
        const float og = sigmoid_f(acc[3][r]);
        const float cv = fg * c[j][r] + ig * gg;
        c[j][r] = cv;
        const float hv = og * tanh_f(cv);
        // C/D layout -> row-major h stage (transpose via LDS for next step's A)
        hs[(r + 8 * h16) * LSTM_H + j * 16 + n16] = (_Float16)hv;
      }
    }
  }

  // ---- final FC: out = h_T @ W_fc^T + b_fc  (two 16x16x32 WMMAs) ----
  asm volatile("s_wait_dscnt 0" ::: "memory");
  v8f oacc;
  {
    const float bo = b_fc[n16];
#pragma unroll
    for (int r = 0; r < 8; ++r) oacc[r] = bo;
  }
#pragma unroll
  for (int kc = 0; kc < 2; ++kc) {
    v16h a;
    {
      const int m  = n16;
      const int k0 = kc * 32 + 8 * h16;
      const v8h lo = *reinterpret_cast<const v8h*>(&hs[m * LSTM_H + k0]);
      const v8h hi = *reinterpret_cast<const v8h*>(&hs[m * LSTM_H + k0 + 16]);
#pragma unroll
      for (int j = 0; j < 8; ++j) { a[j] = lo[j]; a[j + 8] = hi[j]; }
    }
    v16h bf;
#pragma unroll
    for (int j = 0; j < 16; ++j) {
      const int kk = kc * 32 + ((j < 8) ? j : j + 8) + 8 * h16; // always < 64
      bf[j] = (_Float16)W_fc[n16 * LSTM_H + kk];
    }
    oacc = __builtin_amdgcn_wmma_f32_16x16x32_f16(
        false, a, false, bf, (short)0, oacc, false, false);
  }
#pragma unroll
  for (int r = 0; r < 8; ++r)
    out[(size_t)(b0 + r + 8 * h16) * LSTM_O + n16] = oacc[r];
}

extern "C" void kernel_launch(void* const* d_in, const int* in_sizes, int n_in,
                              void* d_out, int out_size, void* d_ws, size_t ws_size,
                              hipStream_t stream) {
  (void)in_sizes; (void)n_in; (void)out_size; (void)d_ws; (void)ws_size;
  const float* x    = (const float*)d_in[0];
  const float* W_ih = (const float*)d_in[1];
  const float* W_hh = (const float*)d_in[2];
  const float* b_ih = (const float*)d_in[3];
  const float* b_hh = (const float*)d_in[4];
  const float* W_fc = (const float*)d_in[5];
  const float* b_fc = (const float*)d_in[6];
  float* out = (float*)d_out;

  const int blocks  = LSTM_B / (ROWS_PER_WAVE * WAVES_PER_WG); // 256
  const int threads = WAVES_PER_WG * 32;                        // 64
  ShallowLSTM_gfx1250_kernel<<<blocks, threads, 0, stream>>>(
      x, W_ih, W_hh, b_ih, b_hh, W_fc, b_fc, out);
}
